// Matrix_Transformer_75213467287643
// MI455X (gfx1250) — matrix operand order repaired and checked by reading, not compiled, not run on hardware
//
#include <hip/hip_runtime.h>
#include <hip/hip_bf16.h>

typedef __attribute__((ext_vector_type(16))) _Float16 v16h;
typedef __attribute__((ext_vector_type(8)))  float    v8f;

union F16Frag { v16h v; _Float16 h[16]; };
union F32x8   { v8f  v; float f[8]; };

#define DIM 512
#define NSEQ 64
#define BATCH 32
#define ROWS 2048          // BATCH*NSEQ
#define MPAIRS 4096        // NSEQ*NSEQ
#define HEADS 8
#define DHEAD 64

// ---------------- ws layout (floats) ----------------
#define OFF_Q     0u
#define OFF_K     1048576u
#define OFF_PSUM  2097152u   // [2][16][512]
#define OFF_PSQ   2113536u   // [2][16][512]
#define OFF_SS    2129920u   // [4][512] scaleq,shiftq,scalek,shiftk
#define OFF_ATTN  2131968u   // [32][8][4096]
#define OFF_PP    0u         // reuse dead q/k region: [32][8][8][512]

// =====================================================================
// Kernel A: q = (subj + tile(semb)) @ Wq^T + bq, k likewise.
// 16x16 WMMA f16 tiles, K=512 in 16 steps of 32. 8 waves/block, 1024 blocks.
// =====================================================================
__global__ __launch_bounds__(256) void qk_gemm_kernel(
    const float* __restrict__ subj, const float* __restrict__ obj,
    const float* __restrict__ semb, const float* __restrict__ oemb,
    const float* __restrict__ Wq,   const float* __restrict__ bq,
    const float* __restrict__ Wk,   const float* __restrict__ bk,
    float* __restrict__ qout, float* __restrict__ kout)
{
    const int wave = threadIdx.x >> 5;
    const int lane = threadIdx.x & 31;
    const int gw   = blockIdx.x * 8 + wave;
    const int prob = gw >> 12;        // 0 = q, 1 = k
    const int tid  = gw & 4095;
    const int mt   = tid >> 5;        // 0..127 row tile
    const int nt   = tid & 31;        // 0..31  col tile

    const float* __restrict__ src  = prob ? obj  : subj;
    const float* __restrict__ emb  = prob ? oemb : semb;
    const float* __restrict__ W    = prob ? Wk   : Wq;
    const float* __restrict__ bias = prob ? bk   : bq;
    float* __restrict__ out        = prob ? kout : qout;

    const int half = lane >> 4;
    const int r16  = lane & 15;
    const int arow = mt * 16 + r16;          // input row (0..2047)
    const int bb   = arow >> 6;              // batch for emb
    const int crow = nt * 16 + r16;          // output channel / W row

    const float* __restrict__ srow = src + (size_t)arow * DIM;
    const float* __restrict__ erow = emb + (size_t)bb * 256;
    const float* __restrict__ wrow = W + (size_t)crow * DIM;

    F32x8 acc; acc.v = (v8f){};
#pragma unroll
    for (int kk = 0; kk < 16; ++kk) {
        const int kbase = kk * 32;
        const int j0 = kbase + half * 16;
        const int j1 = kbase + half * 16 + 8;
        F16Frag a, bfr;
#pragma unroll
        for (int i = 0; i < 8; ++i) {
            const int ja = j0 + i, jb = j1 + i;
            a.h[i]     = (_Float16)(srow[ja] + erow[ja & 255]);
            a.h[i + 8] = (_Float16)(srow[jb] + erow[jb & 255]);
        }
        const int jB = kbase + half * 16;
#pragma unroll
        for (int i = 0; i < 16; ++i) bfr.h[i] = (_Float16)wrow[jB + i];
        acc.v = __builtin_amdgcn_wmma_f32_16x16x32_f16(
            false, a.v, false, bfr.v, (short)0, acc.v, false, false);
    }
    const float bc = bias[crow];
#pragma unroll
    for (int v = 0; v < 8; ++v) {
        const int m = mt * 16 + v + half * 8;
        out[(size_t)m * DIM + crow] = acc.f[v] + bc;
    }
}

// =====================================================================
// Kernel B1: per-channel partial sum / sumsq over 128-row blocks.
// =====================================================================
__global__ __launch_bounds__(256) void bn_partial_kernel(
    const float* __restrict__ q, const float* __restrict__ k,
    float* __restrict__ psum, float* __restrict__ psq)
{
    const int mat = blockIdx.x >> 4;
    const int rb  = blockIdx.x & 15;
    const float* __restrict__ Y = mat ? k : q;
#pragma unroll
    for (int ci = 0; ci < 2; ++ci) {
        const int c = threadIdx.x + ci * 256;
        float s = 0.f, s2 = 0.f;
        for (int r = rb * 128; r < rb * 128 + 128; ++r) {
            const float x = Y[(size_t)r * DIM + c];
            s += x; s2 += x * x;
        }
        psum[((size_t)mat * 16 + rb) * DIM + c] = s;
        psq [((size_t)mat * 16 + rb) * DIM + c] = s2;
    }
}

// =====================================================================
// Kernel B2: finalize BN -> scale = gamma*rsqrt(var+eps), shift = beta-mean*scale
// =====================================================================
__global__ __launch_bounds__(256) void bn_finalize_kernel(
    const float* __restrict__ psum, const float* __restrict__ psq,
    const float* __restrict__ gamma, const float* __restrict__ beta,
    float* __restrict__ ss)
{
    const int id  = blockIdx.x * 256 + threadIdx.x;   // 0..1023
    const int mat = id >> 9;
    const int c   = id & 511;
    float s = 0.f, s2 = 0.f;
#pragma unroll
    for (int rb = 0; rb < 16; ++rb) {
        s  += psum[((size_t)mat * 16 + rb) * DIM + c];
        s2 += psq [((size_t)mat * 16 + rb) * DIM + c];
    }
    const float mean = s * (1.0f / (float)ROWS);
    const float var  = s2 * (1.0f / (float)ROWS) - mean * mean;
    const float inv  = rsqrtf(var + 1e-5f);
    const float scale = gamma[c] * inv;
    const float shift = beta[c] - mean * scale;
    ss[(size_t)(mat * 2 + 0) * DIM + c] = scale;
    ss[(size_t)(mat * 2 + 1) * DIM + c] = shift;
}

// =====================================================================
// Kernel C: per (b,h): scores = BN(q) BN(k)^T * 0.125 via WMMA f16,
// block softmax over flattened 4096, write attn. 256 blocks, 8 waves.
// =====================================================================
__global__ __launch_bounds__(256) void attn_kernel(
    const float* __restrict__ q, const float* __restrict__ k,
    const float* __restrict__ ss, float* __restrict__ attn)
{
    __shared__ float lsq[DHEAD], lhq[DHEAD], lsk[DHEAD], lhk[DHEAD];
    __shared__ float sc[MPAIRS];
    __shared__ float red[256];

    const int b = blockIdx.x >> 3;
    const int h = blockIdx.x & 7;
    const int t = threadIdx.x;
    if (t < 64) {
        const int c = h * DHEAD + t;
        lsq[t] = ss[0 * DIM + c];  lhq[t] = ss[1 * DIM + c];
        lsk[t] = ss[2 * DIM + c];  lhk[t] = ss[3 * DIM + c];
    }
    __syncthreads();

    const int wave = t >> 5, lane = t & 31;
    const int half = lane >> 4, r16 = lane & 15;
    const float* __restrict__ qb = q + (size_t)(b * NSEQ) * DIM + h * DHEAD;
    const float* __restrict__ kb = k + (size_t)(b * NSEQ) * DIM + h * DHEAD;

#pragma unroll
    for (int tt = 0; tt < 2; ++tt) {
        const int tile = wave * 2 + tt;
        const int ti = tile >> 2, tj = tile & 3;
        const int n1 = ti * 16 + r16;     // q row for A frag
        const int n2 = tj * 16 + r16;     // k row for B frag (score column)
        F32x8 acc; acc.v = (v8f){};
#pragma unroll
        for (int kk = 0; kk < 2; ++kk) {
            const int kbase = kk * 32;
            F16Frag a, bf;
            const int d0 = kbase + half * 16;
            const int d1 = kbase + half * 16 + 8;
#pragma unroll
            for (int i = 0; i < 8; ++i) {
                const int da = d0 + i, db = d1 + i;
                a.h[i]     = (_Float16)(qb[(size_t)n1 * DIM + da] * lsq[da] + lhq[da]);
                a.h[i + 8] = (_Float16)(qb[(size_t)n1 * DIM + db] * lsq[db] + lhq[db]);
            }
            const int dB = kbase + half * 16;
#pragma unroll
            for (int i = 0; i < 16; ++i) {
                const int d = dB + i;
                bf.h[i] = (_Float16)(kb[(size_t)n2 * DIM + d] * lsk[d] + lhk[d]);
            }
            acc.v = __builtin_amdgcn_wmma_f32_16x16x32_f16(
                false, a.v, false, bf.v, (short)0, acc.v, false, false);
        }
#pragma unroll
        for (int v = 0; v < 8; ++v) {
            const int m = ti * 16 + v + half * 8;              // score row n1
            sc[m * NSEQ + tj * 16 + r16] = acc.f[v] * 0.125f;  // SCALE = 64^-0.5
        }
    }
    __syncthreads();

    // ---- softmax over 4096 (fixed-order tree => deterministic) ----
    float lm = -3.4e38f;
#pragma unroll
    for (int i = 0; i < 16; ++i) lm = fmaxf(lm, sc[i * 256 + t]);
    red[t] = lm; __syncthreads();
    for (int sN = 128; sN > 0; sN >>= 1) {
        if (t < sN) red[t] = fmaxf(red[t], red[t + sN]);
        __syncthreads();
    }
    const float gm = red[0];
    __syncthreads();
    float lsum = 0.f;
#pragma unroll
    for (int i = 0; i < 16; ++i) {
        const float e = __expf(sc[i * 256 + t] - gm);
        sc[i * 256 + t] = e;
        lsum += e;
    }
    red[t] = lsum; __syncthreads();
    for (int sN = 128; sN > 0; sN >>= 1) {
        if (t < sN) red[t] += red[t + sN];
        __syncthreads();
    }
    const float inv = 1.0f / red[0];
    float* __restrict__ ap = attn + (size_t)(b * HEADS + h) * MPAIRS;
#pragma unroll
    for (int i = 0; i < 16; ++i) ap[i * 256 + t] = sc[i * 256 + t] * inv;
}

// =====================================================================
// Kernel D (HBM-bound): partial pooled[b,h,j] over 512-row m-chunks.
// 256 blocks: b(32) x mchunk(8). Each thread owns 2 channels (float2
// global loads, 256B/instr/wave); attn weights broadcast from LDS as
// float4 per head per 4-row group (ds_load_b128).
// =====================================================================
__global__ __launch_bounds__(256) void pool_partial_kernel(
    const float* __restrict__ sod, const float* __restrict__ attn,
    float* __restrict__ pp)
{
    __shared__ float at[HEADS][512];
    const int b  = blockIdx.x >> 3;
    const int mc = blockIdx.x & 7;
    const int t  = threadIdx.x;

#pragma unroll
    for (int z = 0; z < 16; ++z) {
        const int idx = z * 256 + t;        // 0..4095
        const int hh = idx >> 9, ii = idx & 511;
        at[hh][ii] = attn[(size_t)(b * HEADS + hh) * MPAIRS + mc * 512 + ii];
    }
    __syncthreads();

    float accx[HEADS], accy[HEADS];
#pragma unroll
    for (int hh = 0; hh < HEADS; ++hh) { accx[hh] = 0.f; accy[hh] = 0.f; }

    // base of this block's m-chunk; thread t covers channels {2t, 2t+1}
    const float2* __restrict__ xp =
        (const float2*)(sod + ((size_t)b * MPAIRS + (size_t)mc * 512) * DIM) + t;
    for (int m4 = 0; m4 < 512; m4 += 4) {
        const float2 x0 = xp[(size_t)(m4 + 0) * 256];
        const float2 x1 = xp[(size_t)(m4 + 1) * 256];
        const float2 x2 = xp[(size_t)(m4 + 2) * 256];
        const float2 x3 = xp[(size_t)(m4 + 3) * 256];
#pragma unroll
        for (int hh = 0; hh < HEADS; ++hh) {
            const float4 a = *(const float4*)&at[hh][m4];   // lane-uniform b128
            accx[hh] += a.x * x0.x + a.y * x1.x + a.z * x2.x + a.w * x3.x;
            accy[hh] += a.x * x0.y + a.y * x1.y + a.z * x2.y + a.w * x3.y;
        }
    }
#pragma unroll
    for (int hh = 0; hh < HEADS; ++hh) {
        float2* __restrict__ dst = (float2*)
            (pp + (((size_t)b * 8 + mc) * HEADS + hh) * DIM) + t;
        *dst = make_float2(accx[hh], accy[hh]);
    }
}

// =====================================================================
// Kernel E: pooled = sum(partials) + tile(delta_emb); out = pooled @ Wv^T + bv
// 32 blocks (one per batch); float4 GEMV.
// =====================================================================
__global__ __launch_bounds__(256) void finalize_out_kernel(
    const float* __restrict__ pp, const float* __restrict__ demb,
    const float* __restrict__ Wv, const float* __restrict__ bv,
    float* __restrict__ out)
{
    __shared__ float pool[HEADS][DIM];
    const int b = blockIdx.x;
    const int t = threadIdx.x;
#pragma unroll
    for (int z = 0; z < 16; ++z) {
        const int idx = z * 256 + t;
        const int hh = idx >> 9, j = idx & 511;
        float s = demb[(size_t)b * 256 + (j & 255)];   // softmax weights sum to 1
#pragma unroll
        for (int mc = 0; mc < 8; ++mc)
            s += pp[(((size_t)b * 8 + mc) * HEADS + hh) * DIM + j];
        pool[hh][j] = s;
    }
    __syncthreads();
#pragma unroll
    for (int ci = 0; ci < 2; ++ci) {
        const int c = t + ci * 256;
        const int hh = c >> 6;
        float acc = bv[c];
        const float4* __restrict__ wr = (const float4*)(Wv + (size_t)c * DIM);
        const float4* __restrict__ pl = (const float4*)&pool[hh][0];
        for (int j4 = 0; j4 < 128; ++j4) {
            const float4 w = wr[j4];
            const float4 p = pl[j4];
            acc += w.x * p.x + w.y * p.y + w.z * p.z + w.w * p.w;
        }
        out[(size_t)b * DIM + c] = acc;
    }
}

// =====================================================================
extern "C" void kernel_launch(void* const* d_in, const int* in_sizes, int n_in,
                              void* d_out, int out_size, void* d_ws, size_t ws_size,
                              hipStream_t stream) {
    (void)in_sizes; (void)n_in; (void)out_size; (void)ws_size;
    const float* subj  = (const float*)d_in[0];
    const float* obj   = (const float*)d_in[1];
    const float* sod   = (const float*)d_in[2];
    const float* semb  = (const float*)d_in[3];
    const float* oemb  = (const float*)d_in[4];
    const float* demb  = (const float*)d_in[5];
    const float* Wq    = (const float*)d_in[6];
    const float* bq    = (const float*)d_in[7];
    const float* Wk    = (const float*)d_in[8];
    const float* bk    = (const float*)d_in[9];
    const float* Wv    = (const float*)d_in[10];
    const float* bv    = (const float*)d_in[11];
    const float* gamma = (const float*)d_in[12];
    const float* beta  = (const float*)d_in[13];
    float* out = (float*)d_out;
    float* ws  = (float*)d_ws;

    qk_gemm_kernel<<<1024, 256, 0, stream>>>(
        subj, obj, semb, oemb, Wq, bq, Wk, bk, ws + OFF_Q, ws + OFF_K);
    bn_partial_kernel<<<32, 256, 0, stream>>>(
        ws + OFF_Q, ws + OFF_K, ws + OFF_PSUM, ws + OFF_PSQ);
    bn_finalize_kernel<<<4, 256, 0, stream>>>(
        ws + OFF_PSUM, ws + OFF_PSQ, gamma, beta, ws + OFF_SS);
    attn_kernel<<<256, 256, 0, stream>>>(
        ws + OFF_Q, ws + OFF_K, ws + OFF_SS, ws + OFF_ATTN);
    pool_partial_kernel<<<256, 256, 0, stream>>>(
        sod, ws + OFF_ATTN, ws + OFF_PP);        // PP reuses the dead q/k region
    finalize_out_kernel<<<32, 256, 0, stream>>>(
        ws + OFF_PP, demb, Wv, bv, out);
}
